// LongcatFlashDecoderLayer_43963285242583
// MI455X (gfx1250) — compile-verified, matrix operand order checked by reading, not run on hardware
//
#include <hip/hip_runtime.h>
#include <cstddef>
#include <cstdint>

typedef _Float16 f16;
typedef __attribute__((ext_vector_type(16))) _Float16 v16h;
typedef __attribute__((ext_vector_type(8)))  _Float16 h8;
typedef __attribute__((ext_vector_type(8)))  float    v8f;

#define TOKENS 2048
#define DMODEL 2048
#define SEQ    1024
#define NHEAD  16
#define QKDIM  192
#define NOPE_D 128
#define ROPE_DIM 64
#define VHEAD  128
#define QLORA  1024
#define KVLORA 512
#define KCAT   576          // KVLORA + ROPE_DIM, fused score K
#define FFN_D  8192
#define EFFN_D 1024
#define NEXP   8
#define NROUTE 10

__device__ __forceinline__ float silu_f(float x) { return x / (1.0f + __expf(-x)); }

// CDNA5 async global->LDS copy (ASYNCcnt-tracked DMA path), 16B per lane.
// VDST = per-lane LDS byte address, VADDR = per-lane 64-bit global address.
__device__ __forceinline__ void async_b128(unsigned lds_off, const void* gaddr) {
    asm volatile("global_load_async_to_lds_b128 %0, %1, off"
                 :: "v"(lds_off), "v"(gaddr) : "memory");
}
__device__ __forceinline__ void wait_async0() {
    asm volatile("s_wait_asynccnt 0x0" ::: "memory");
}

// ---------------------------------------------------------------------------
// WMMA GEMM with double-buffered async-DMA pipeline.
//   C[M,N] = epilogue( alpha * A[M,K] * op(B) )
//   A f16 [M,K] row-major, M multiple of 128 (no guards; async DMA to LDS)
//   TRANSB=1 : B [N,K] row-major;  B16 ? f16 (async DMA) : f32 weights (VALU cvt)
//   TRANSB=0 : B [K,N] row-major f16 (transposed VALU staging)
//   C16=1    : store f16 (no accumulate); C16=0 : f32 with optional accumulate
// Pipeline per K-step: wait_asynccnt 0 ; barrier ; stage(t+1) ; compute(t)
// Tiles: BM=128 BN=128 BK=32, 256 threads = 8 waves, wave tile 64x32.
// ---------------------------------------------------------------------------
template <int TRANSB, int B16, int C16>
__global__ __launch_bounds__(256) void gemm_f16_wmma(
    const f16* __restrict__ A, const void* __restrict__ Bv, float* __restrict__ C,
    int M, int N, int K, int lda, int ldb, int ldc, float alpha,
    const float* __restrict__ rowScale, const float* __restrict__ gateC, int accumulate)
{
    __shared__ __align__(16) f16 sA[2][128 * 40];
    __shared__ __align__(16) f16 sB[2][128 * 40];
    const float* Bf = (const float*)Bv;
    const f16*  Bh  = (const f16*)Bv;

    const int tid  = threadIdx.x;
    const int m0   = blockIdx.y * 128;
    const int n0   = blockIdx.x * 128;
    const int wid  = tid >> 5;
    const int lane = tid & 31;
    const int wm   = wid >> 2;        // 0..1  (64 rows each)
    const int wn   = wid & 3;         // 0..3  (32 cols each)
    const int hsel = lane >> 4;       // K-half select per WMMA 16-bit layout
    const int lrow = lane & 15;

    v8f acc[4][2];
#pragma unroll
    for (int mi = 0; mi < 4; ++mi)
#pragma unroll
        for (int ni = 0; ni < 2; ++ni)
#pragma unroll
            for (int j = 0; j < 8; ++j) acc[mi][ni][j] = 0.0f;

    const int nk = K >> 5;

    auto stage = [&](int t) {
        const int k0 = t << 5;
        f16* dA = sA[t & 1];
        f16* dB = sB[t & 1];
        // A tile: async DMA straight into LDS (512 x 16B chunks)
#pragma unroll
        for (int i = 0; i < 2; ++i) {
            int ch = i * 256 + tid;
            int r = ch >> 2, qc = (ch & 3) * 8;
            async_b128((unsigned)(size_t)&dA[r * 40 + qc],
                       A + (size_t)(m0 + r) * lda + k0 + qc);
        }
        if (TRANSB && B16) {
#pragma unroll
            for (int i = 0; i < 2; ++i) {
                int ch = i * 256 + tid;
                int r = ch >> 2, qc = (ch & 3) * 8;
                if (n0 + r < N) {
                    async_b128((unsigned)(size_t)&dB[r * 40 + qc],
                               Bh + (size_t)(n0 + r) * ldb + k0 + qc);
                } else {
                    f16* d = &dB[r * 40 + qc];
#pragma unroll
                    for (int j = 0; j < 8; ++j) d[j] = (f16)0.f;
                }
            }
        } else if (TRANSB) {
            // f32 weights -> f16 convert while staging
#pragma unroll
            for (int i = 0; i < 4; ++i) {
                int e = (i * 256 + tid) * 4;
                int r = e >> 5, c = e & 31;
                f16 h0 = (f16)0.f, h1 = (f16)0.f, h2 = (f16)0.f, h3 = (f16)0.f;
                if (n0 + r < N) {
                    const float* p = Bf + (size_t)(n0 + r) * ldb + k0 + c;
                    h0 = (f16)p[0]; h1 = (f16)p[1]; h2 = (f16)p[2]; h3 = (f16)p[3];
                    if (k0 + 32 < K) __builtin_prefetch(p + 32, 0, 0);
                }
                f16* d = &dB[r * 40 + c];
                d[0] = h0; d[1] = h1; d[2] = h2; d[3] = h3;
            }
        } else {
            // B16 [K,N] -> [n][k] transposed staging
#pragma unroll
            for (int i = 0; i < 4; ++i) {
                int e  = (i * 256 + tid) * 4;
                int kk = e >> 7, nn = e & 127;
                const f16* pb = Bh + (size_t)(k0 + kk) * ldb + n0 + nn;
#pragma unroll
                for (int j = 0; j < 4; ++j) {
                    f16 hv = (f16)0.f;
                    if (n0 + nn + j < N) hv = pb[j];
                    dB[(nn + j) * 40 + kk] = hv;
                }
            }
        }
    };

    stage(0);
    for (int t = 0; t < nk; ++t) {
        wait_async0();          // tile t async copies have landed in LDS
        __syncthreads();        // all waves' staging of tile t visible; readers of
                                // buffer (t+1)&1 from tile t-1 are done
        if (t + 1 < nk) stage(t + 1);   // DMA + weight-cvt overlap with compute(t)

        const f16* cA = sA[t & 1];
        const f16* cB = sB[t & 1];
        v16h af[4], bf[2];
#pragma unroll
        for (int mi = 0; mi < 4; ++mi) {
            int r = wm * 64 + mi * 16 + lrow;
            const h8* p0 = (const h8*)(&cA[r * 40 + hsel * 8]);
            const h8* p1 = (const h8*)(&cA[r * 40 + hsel * 8 + 16]);
            af[mi] = __builtin_shufflevector(*p0, *p1, 0,1,2,3,4,5,6,7,8,9,10,11,12,13,14,15);
        }
#pragma unroll
        for (int ni = 0; ni < 2; ++ni) {
            int r = wn * 32 + ni * 16 + lrow;
            const h8* p0 = (const h8*)(&cB[r * 40 + hsel * 8]);
            const h8* p1 = (const h8*)(&cB[r * 40 + hsel * 8 + 16]);
            bf[ni] = __builtin_shufflevector(*p0, *p1, 0,1,2,3,4,5,6,7,8,9,10,11,12,13,14,15);
        }
#pragma unroll
        for (int mi = 0; mi < 4; ++mi)
#pragma unroll
            for (int ni = 0; ni < 2; ++ni)
                acc[mi][ni] = __builtin_amdgcn_wmma_f32_16x16x32_f16(
                    false, af[mi], false, bf[ni], (short)0, acc[mi][ni], false, false);
    }

    // ---- epilogue
#pragma unroll
    for (int mi = 0; mi < 4; ++mi) {
#pragma unroll
        for (int ni = 0; ni < 2; ++ni) {
            int col = n0 + wn * 32 + ni * 16 + lrow;
            if (col < N) {
#pragma unroll
                for (int j = 0; j < 8; ++j) {
                    int row = m0 + wm * 64 + mi * 16 + hsel * 8 + j;
                    if (row < M) {
                        size_t idx = (size_t)row * ldc + col;
                        float v = acc[mi][ni][j] * alpha;
                        if (gateC)    v *= silu_f(gateC[idx]);
                        if (rowScale) v *= rowScale[row];
                        if (C16) {
                            ((f16*)C)[idx] = (f16)v;
                        } else {
                            if (accumulate) v += C[idx];
                            C[idx] = v;
                        }
                    }
                }
            }
        }
    }
}

// ---------------------------------------------------------------------------
// RMSNorm: one block per row; input f32 or f16, output f16.
// ---------------------------------------------------------------------------
template <int IN16>
__global__ __launch_bounds__(256) void rms_kernel(
    const void* __restrict__ xv, const float* __restrict__ w, f16* __restrict__ y,
    int cols, int in_ld, int out_ld, float scale)
{
    __shared__ float red[256];
    int row = blockIdx.x, tid = threadIdx.x;
    const float* xf = (const float*)xv;
    const f16*   xh = (const f16*)xv;
    float s = 0.f;
    for (int c = tid; c < cols; c += 256) {
        float v = IN16 ? (float)xh[(size_t)row * in_ld + c] : xf[(size_t)row * in_ld + c];
        s += v * v;
    }
    red[tid] = s; __syncthreads();
    for (int st = 128; st > 0; st >>= 1) { if (tid < st) red[tid] += red[tid + st]; __syncthreads(); }
    float inv = rsqrtf(red[0] / (float)cols + 1e-5f) * scale;
    __syncthreads();
    for (int c = tid; c < cols; c += 256) {
        float v = IN16 ? (float)xh[(size_t)row * in_ld + c] : xf[(size_t)row * in_ld + c];
        y[(size_t)row * out_ld + c] = (f16)(v * inv * w[c]);
    }
}

// ---------------------------------------------------------------------------
// RoPE for q_pe: qb16[t, h*192+128..) -> qcat[b,h][l, 512..576), scaled by SQ.
// qcat layout: [(b*16+h)*1024 + l][576]
// ---------------------------------------------------------------------------
__global__ void rope_q_kernel(const f16* __restrict__ q, f16* __restrict__ qcat)
{
    int idx = blockIdx.x * blockDim.x + threadIdx.x;     // (t,h,j)
    int j = idx & 31;
    int h = (idx >> 5) & 15;
    int t = idx >> 9;
    if (t >= TOKENS) return;
    int l = t & (SEQ - 1);
    int b = t >> 10;
    float inv = __powf(10000.0f, -(float)(2 * j) / 64.0f);
    float sv, cv; __sincosf((float)l * inv, &sv, &cv);
    const f16* p = q + (size_t)t * (NHEAD * QKDIM) + h * QKDIM + NOPE_D + 2 * j;
    float x1 = (float)p[0], x2 = (float)p[1];
    f16* o = qcat + ((size_t)(b * NHEAD + h) * SEQ + l) * KCAT + KVLORA + 2 * j;
    const float SQv = 1.4142135623730951f;
    o[0] = (f16)((x1 * cv - x2 * sv) * SQv);
    o[1] = (f16)((x1 * sv + x2 * cv) * SQv);
}

// RoPE for k_pe: ckv16[t, 512..576) -> kcat[t][512..576)
__global__ void rope_k_kernel(const f16* __restrict__ ckv, f16* __restrict__ kcat)
{
    int idx = blockIdx.x * blockDim.x + threadIdx.x;     // (t,j)
    int j = idx & 31;
    int t = idx >> 5;
    if (t >= TOKENS) return;
    int l = t & (SEQ - 1);
    float inv = __powf(10000.0f, -(float)(2 * j) / 64.0f);
    float sv, cv; __sincosf((float)l * inv, &sv, &cv);
    const f16* p = ckv + (size_t)t * KCAT + KVLORA + 2 * j;
    float x1 = (float)p[0], x2 = (float)p[1];
    f16* o = kcat + (size_t)t * KCAT + KVLORA + 2 * j;
    o[0] = (f16)(x1 * cv - x2 * sv);
    o[1] = (f16)(x1 * sv + x2 * cv);
}

// ---------------------------------------------------------------------------
// Causal softmax over [SEQ x SEQ] f32 scores; emits f16 probabilities.
// ---------------------------------------------------------------------------
__global__ __launch_bounds__(256) void softmax_causal_kernel(
    float* __restrict__ s, f16* __restrict__ p, int n)
{
    __shared__ float red[256];
    int row = blockIdx.x, tid = threadIdx.x;
    float* r = s + (size_t)row * n;
    f16*   pr = p + (size_t)row * n;
    int valid = row + 1;
    float mx = -1e30f;
    for (int c = tid; c < valid; c += 256) mx = fmaxf(mx, r[c]);
    red[tid] = mx; __syncthreads();
    for (int st = 128; st > 0; st >>= 1) { if (tid < st) red[tid] = fmaxf(red[tid], red[tid + st]); __syncthreads(); }
    mx = red[0]; __syncthreads();
    float sum = 0.f;
    for (int c = tid; c < valid; c += 256) { float e = __expf(r[c] - mx); r[c] = e; sum += e; }
    red[tid] = sum; __syncthreads();
    for (int st = 128; st > 0; st >>= 1) { if (tid < st) red[tid] += red[tid + st]; __syncthreads(); }
    float invs = 1.0f / red[0];
    __syncthreads();
    for (int c = tid; c < valid; c += 256) pr[c] = (f16)(r[c] * invs);
    for (int c = valid + tid; c < n; c += 256) pr[c] = (f16)0.f;
}

// ---------------------------------------------------------------------------
// MoE routing: softmax over 10 logits, top-2 of (p+bias), fill ce[e][t], idw[t]
// ---------------------------------------------------------------------------
__global__ void route_kernel(const float* __restrict__ rs, const float* __restrict__ ebias,
                             float* __restrict__ ce, float* __restrict__ idw, int T)
{
    int t = blockIdx.x * blockDim.x + threadIdx.x;
    if (t >= T) return;
    float v[NROUTE];
    float mx = -1e30f;
    for (int e = 0; e < NROUTE; ++e) { v[e] = rs[(size_t)t * NROUTE + e]; mx = fmaxf(mx, v[e]); }
    float sum = 0.f;
    for (int e = 0; e < NROUTE; ++e) { v[e] = __expf(v[e] - mx); sum += v[e]; }
    float invs = 1.0f / sum;
    for (int e = 0; e < NROUTE; ++e) v[e] *= invs;
    int i1 = 0; float b1 = -1e30f;
    for (int e = 0; e < NROUTE; ++e) { float bv = v[e] + ebias[e]; if (bv > b1) { b1 = bv; i1 = e; } }
    int i2 = 0; float b2 = -1e30f;
    for (int e = 0; e < NROUTE; ++e) { if (e == i1) continue; float bv = v[e] + ebias[e]; if (bv > b2) { b2 = bv; i2 = e; } }
    float cloc[NEXP];
    for (int e = 0; e < NEXP; ++e) cloc[e] = 0.f;
    float id = 0.f;
    int sel[2] = { i1, i2 };
    for (int k = 0; k < 2; ++k) {
        int e = sel[k];
        float w = v[e] * 1.5f;
        if (e >= NEXP) id += w; else cloc[e] += w;
    }
    for (int e = 0; e < NEXP; ++e) ce[(size_t)e * T + t] = cloc[e];
    idw[t] = id;
}

__global__ void scale_rows_kernel(const f16* __restrict__ x, const float* __restrict__ s,
                                  float* __restrict__ y, int cols)
{
    int row = blockIdx.y;
    int c = blockIdx.x * blockDim.x + threadIdx.x;
    size_t idx = (size_t)row * cols + c;
    y[idx] = (float)x[idx] * s[row];
}

__global__ void add2_kernel(const float* __restrict__ a, const float* __restrict__ b,
                            float* __restrict__ o, size_t n)
{
    size_t i = (size_t)blockIdx.x * blockDim.x + threadIdx.x;
    if (i < n) o[i] = a[i] + b[i];
}

// ---------------------------------------------------------------------------
template <int TRANSB, int B16, int C16>
static void gemm_launch(const f16* A, const void* B, void* C,
                        int M, int N, int K, int lda, int ldb, int ldc, float alpha,
                        const float* rowScale, const float* gateC, int accumulate,
                        hipStream_t st)
{
    dim3 grid((N + 127) / 128, (M + 127) / 128);
    gemm_f16_wmma<TRANSB, B16, C16><<<grid, 256, 0, st>>>(
        A, B, (float*)C, M, N, K, lda, ldb, ldc, alpha, rowScale, gateC, accumulate);
}

extern "C" void kernel_launch(void* const* d_in, const int* in_sizes, int n_in,
                              void* d_out, int out_size, void* d_ws, size_t ws_size,
                              hipStream_t stream)
{
    (void)in_sizes; (void)n_in; (void)out_size; (void)ws_size;
    const float* X = (const float*)d_in[0];
    const float* in_ln[2]   = { (const float*)d_in[2],  (const float*)d_in[15] };
    const float* post_ln[2] = { (const float*)d_in[3],  (const float*)d_in[16] };
    const float* qa_w[2]    = { (const float*)d_in[4],  (const float*)d_in[17] };
    const float* qa_ln[2]   = { (const float*)d_in[5],  (const float*)d_in[18] };
    const float* qb_w[2]    = { (const float*)d_in[6],  (const float*)d_in[19] };
    const float* kva_w[2]   = { (const float*)d_in[7],  (const float*)d_in[20] };
    const float* kva_ln[2]  = { (const float*)d_in[8],  (const float*)d_in[21] };
    const float* embq[2]    = { (const float*)d_in[9],  (const float*)d_in[22] };
    const float* unemb[2]   = { (const float*)d_in[10], (const float*)d_in[23] };
    const float* o_w[2]     = { (const float*)d_in[11], (const float*)d_in[24] };
    const float* gate_w[2]  = { (const float*)d_in[12], (const float*)d_in[25] };
    const float* up_w[2]    = { (const float*)d_in[13], (const float*)d_in[26] };
    const float* down_w[2]  = { (const float*)d_in[14], (const float*)d_in[27] };
    const float* router_w   = (const float*)d_in[28];
    const float* e_bias     = (const float*)d_in[29];
    const float* e_gate     = (const float*)d_in[30];
    const float* e_up       = (const float*)d_in[31];
    const float* e_down     = (const float*)d_in[32];

    // workspace bump allocation (~145 MB): f32 residual stream + f16 activations
    char* base = (char*)d_ws;
    size_t off = 0;
    auto allocf = [&](size_t n) { float* p = (float*)(base + off); off += ((n * 4 + 255) & ~(size_t)255); return p; };
    auto alloch = [&](size_t n) { f16*   p = (f16*)(base + off);   off += ((n * 2 + 255) & ~(size_t)255); return p; };

    float* hcur     = allocf((size_t)TOKENS * DMODEL);
    float* sbuf     = allocf((size_t)SEQ * SEQ);
    float* buf1     = allocf((size_t)TOKENS * 2048);        // gate pre-activation (f32)
    float* shortcut = allocf((size_t)TOKENS * DMODEL);
    float* rsc      = allocf((size_t)TOKENS * NROUTE);
    float* ce       = allocf((size_t)NEXP * TOKENS);
    float* idw      = allocf((size_t)TOKENS);

    f16* hn16   = alloch((size_t)TOKENS * DMODEL);
    f16* qa16   = alloch((size_t)TOKENS * QLORA);
    f16* qb16   = alloch((size_t)TOKENS * NHEAD * QKDIM);
    f16* ckv16  = alloch((size_t)TOKENS * KCAT);
    f16* qcat16 = alloch((size_t)2 * NHEAD * SEQ * KCAT);   // [SQ*q_abs | SQ*q_pe_rot]
    f16* kcat16 = alloch((size_t)TOKENS * KCAT);            // [SKV*rms(kv) | k_pe_rot]
    f16* p16    = alloch((size_t)SEQ * SEQ);
    f16* olat16 = alloch((size_t)SEQ * KVLORA);
    f16* attn16 = alloch((size_t)TOKENS * NHEAD * VHEAD);
    f16* buf16  = alloch((size_t)TOKENS * 2048);

    const float SCALE = 0.07216878364870323f;  // 192^-0.5
    const float SQv   = 1.4142135623730951f;   // sqrt(2048/1024)
    const float SKVv  = 2.0f;                  // sqrt(2048/512)

    hipMemcpyAsync(hcur, X, sizeof(float) * (size_t)TOKENS * DMODEL,
                   hipMemcpyDeviceToDevice, stream);

    for (int i = 0; i < 2; ++i) {
        // hn16 = rms(h, in_ln)
        rms_kernel<0><<<TOKENS, 256, 0, stream>>>(hcur, in_ln[i], hn16, DMODEL, DMODEL, DMODEL, 1.0f);
        // q low-rank path
        gemm_launch<1,0,1>(hn16, qa_w[i], qa16, TOKENS, QLORA, DMODEL, DMODEL, DMODEL, QLORA, 1.0f, nullptr, nullptr, 0, stream);
        rms_kernel<1><<<TOKENS, 256, 0, stream>>>(qa16, qa_ln[i], qa16, QLORA, QLORA, QLORA, 1.0f);
        gemm_launch<1,0,1>(qa16, qb_w[i], qb16, TOKENS, NHEAD * QKDIM, QLORA, QLORA, QLORA, NHEAD * QKDIM, 1.0f, nullptr, nullptr, 0, stream);
        // kv low-rank path: ckv16 = hn @ kva_w^T; kcat = [SKV*rms(kv) | rope(k_pe)]
        gemm_launch<1,0,1>(hn16, kva_w[i], ckv16, TOKENS, KCAT, DMODEL, DMODEL, DMODEL, KCAT, 1.0f, nullptr, nullptr, 0, stream);
        rms_kernel<1><<<TOKENS, 256, 0, stream>>>(ckv16, kva_ln[i], kcat16, KVLORA, KCAT, KCAT, SKVv);
        rope_k_kernel<<<(TOKENS * ROPE_DIM / 2) / 256, 256, 0, stream>>>(ckv16, kcat16);
        rope_q_kernel<<<(TOKENS * NHEAD * ROPE_DIM / 2) / 256, 256, 0, stream>>>(qb16, qcat16);

        // attention per (batch, head) with low-rank absorption + fused nope|pe scores
        for (int b = 0; b < 2; ++b) {
            for (int h = 0; h < NHEAD; ++h) {
                const f16* qn   = qb16 + (size_t)b * SEQ * (NHEAD * QKDIM) + (size_t)h * QKDIM;
                f16* qcat_bh    = qcat16 + (size_t)(b * NHEAD + h) * SEQ * KCAT;
                const f16* kcat_b = kcat16 + (size_t)b * SEQ * KCAT;
                // qcat[:, 0:512] = SQ * q_nope @ embq[h]^T
                gemm_launch<1,0,1>(qn, embq[i] + (size_t)h * KVLORA * NOPE_D, qcat_bh,
                                   SEQ, KVLORA, NOPE_D, NHEAD * QKDIM, NOPE_D, KCAT,
                                   SQv, nullptr, nullptr, 0, stream);
                // scores = SCALE * qcat @ kcat[b]^T   (single K=576 f16 GEMM)
                gemm_launch<1,1,0>(qcat_bh, kcat_b, sbuf,
                                   SEQ, SEQ, KCAT, KCAT, KCAT, SEQ, SCALE, nullptr, nullptr, 0, stream);
                softmax_causal_kernel<<<SEQ, 256, 0, stream>>>(sbuf, p16, SEQ);
                // o_lat = P @ kv_part(kcat[b])        [SEQ, KVLORA], B is [K,N] ldb=576
                gemm_launch<0,1,1>(p16, kcat_b, olat16,
                                   SEQ, KVLORA, SEQ, SEQ, KCAT, KVLORA, 1.0f, nullptr, nullptr, 0, stream);
                // attn[b,:,h*128:] = o_lat @ unemb[h]^T
                gemm_launch<1,0,1>(olat16, unemb[i] + (size_t)h * VHEAD * KVLORA,
                                   attn16 + (size_t)b * SEQ * (NHEAD * VHEAD) + (size_t)h * VHEAD,
                                   SEQ, VHEAD, KVLORA, KVLORA, KVLORA, NHEAD * VHEAD, 1.0f, nullptr, nullptr, 0, stream);
            }
        }
        // h += attn @ o_w^T  (residual fused, f32 accumulate)
        gemm_launch<1,0,0>(attn16, o_w[i], hcur, TOKENS, DMODEL, NHEAD * VHEAD,
                           NHEAD * VHEAD, NHEAD * VHEAD, DMODEL, 1.0f, nullptr, nullptr, 1, stream);

        // hn16 = rms(h, post_ln)
        rms_kernel<0><<<TOKENS, 256, 0, stream>>>(hcur, post_ln[i], hn16, DMODEL, DMODEL, DMODEL, 1.0f);

        if (i == 0) {
            // MoE shortcut (dense over all experts, as in reference)
            gemm_launch<1,0,0>(hn16, router_w, rsc, TOKENS, NROUTE, DMODEL, DMODEL, DMODEL, NROUTE, 1.0f, nullptr, nullptr, 0, stream);
            route_kernel<<<TOKENS / 256, 256, 0, stream>>>(rsc, e_bias, ce, idw, TOKENS);
            scale_rows_kernel<<<dim3(DMODEL / 256, TOKENS), 256, 0, stream>>>(hn16, idw, shortcut, DMODEL);
            for (int e = 0; e < NEXP; ++e) {
                gemm_launch<1,0,0>(hn16, e_gate + (size_t)e * EFFN_D * DMODEL, buf1,
                                   TOKENS, EFFN_D, DMODEL, DMODEL, DMODEL, EFFN_D, 1.0f, nullptr, nullptr, 0, stream);
                gemm_launch<1,0,1>(hn16, e_up + (size_t)e * EFFN_D * DMODEL, buf16,
                                   TOKENS, EFFN_D, DMODEL, DMODEL, DMODEL, EFFN_D, 1.0f, nullptr, buf1, 0, stream);
                gemm_launch<1,0,0>(buf16, e_down + (size_t)e * DMODEL * EFFN_D, shortcut,
                                   TOKENS, DMODEL, EFFN_D, EFFN_D, EFFN_D, DMODEL, 1.0f,
                                   ce + (size_t)e * TOKENS, nullptr, 1, stream);
            }
        }

        // dense FFN, chunked over FFN dim (4 x 2048) to bound workspace
        for (int c = 0; c < 4; ++c) {
            const float* gptr = gate_w[i] + (size_t)c * 2048 * DMODEL;
            const float* uptr = up_w[i]   + (size_t)c * 2048 * DMODEL;
            const float* dptr = down_w[i] + (size_t)c * 2048;   // column slice, ldb = FFN_D
            gemm_launch<1,0,0>(hn16, gptr, buf1, TOKENS, 2048, DMODEL, DMODEL, DMODEL, 2048, 1.0f, nullptr, nullptr, 0, stream);
            gemm_launch<1,0,1>(hn16, uptr, buf16, TOKENS, 2048, DMODEL, DMODEL, DMODEL, 2048, 1.0f, nullptr, buf1, 0, stream);
            gemm_launch<1,0,0>(buf16, dptr, hcur, TOKENS, DMODEL, 2048, 2048, FFN_D, DMODEL, 1.0f, nullptr, nullptr, 1, stream);
        }
    }

    add2_kernel<<<(TOKENS * DMODEL) / 256, 256, 0, stream>>>(hcur, shortcut, (float*)d_out, (size_t)TOKENS * DMODEL);
}
